// OFTCrossAttnProcessor_12403865551104
// MI455X (gfx1250) — matrix operand order repaired and checked by reading, not compiled, not run on hardware
//
#include <hip/hip_runtime.h>

typedef __attribute__((ext_vector_type(16))) __bf16 v16bf;
typedef __attribute__((ext_vector_type(8)))  float  v8f;

#define H_DIM   1280
#define NBLK    8
#define BSZ     160
#define BPAD    256        // padded column pitch for rotation blocks
#define NHEADS  20
#define HD      64
#define BATCH   16
#define SEQ     1024
#define SEQE    77
#define CENC    768
#define CONSTRAINT_V 1.28f
#define EPS_N   1e-8f

struct alignas(16) U4 { unsigned x, y, z, w; };
static __device__ __forceinline__ U4 zero4() { U4 z; z.x = z.y = z.z = z.w = 0u; return z; }

static __device__ __forceinline__ v8f wmma_bf16(v16bf a, v16bf b, v8f c) {
  return __builtin_amdgcn_wmma_f32_16x16x32_bf16(false, a, false, b, (short)0, c, false, false);
}

// ---------------------------------------------------------------- utilities
__global__ void zero_f32_kernel(float* p, int n) {
  int i = blockIdx.x * blockDim.x + threadIdx.x;
  if (i < n) p[i] = 0.0f;
}

struct alignas(16) F4 { float x, y, z, w; };

__global__ void cvt_f32_bf16_kernel(const float* __restrict__ in,
                                    __bf16* __restrict__ out, int n4) {
  for (int i = blockIdx.x * blockDim.x + threadIdx.x; i < n4;
       i += gridDim.x * blockDim.x) {
    F4 f = ((const F4*)in)[i];
    union { unsigned long long u; __bf16 h[4]; } o;
    o.h[0] = (__bf16)f.x; o.h[1] = (__bf16)f.y;
    o.h[2] = (__bf16)f.z; o.h[3] = (__bf16)f.w;
    ((unsigned long long*)out)[i] = o.u;
  }
}

// ------------------------------------------------- Cayley: skew + global norm
__global__ __launch_bounds__(256)
void skew_norm_kernel(const float* __restrict__ W, float* __restrict__ Q,
                      float* __restrict__ sumsq) {
  const int r = blockIdx.x;
  const int tid = threadIdx.x;
  const float* Wr = W + (long)r * BSZ * BSZ;
  float* Qr = Q + (long)r * BSZ * BSZ;
  float local = 0.0f;
  for (int idx = tid; idx < BSZ * BSZ; idx += 256) {
    int i = idx / BSZ, j = idx % BSZ;
    float q = Wr[i * BSZ + j] - Wr[j * BSZ + i];
    Qr[idx] = q;
    local += q * q;
  }
  __shared__ float red[256];
  red[tid] = local;
  __syncthreads();
  for (int s = 128; s > 0; s >>= 1) {
    if (tid < s) red[tid] += red[tid + s];
    __syncthreads();
  }
  if (tid == 0) atomicAdd(sumsq, red[0]);
}

// --------------------------- Cayley: Gauss-Jordan solve (I-Q)X = (I+Q) in LDS
// Emits zero-padded 160x256 bf16 blocks so downstream GEMM B-loads are always
// 16B vector loads (no scalar edge path needed).
__global__ __launch_bounds__(256)
void cayley_kernel(const float* __restrict__ Q, const float* __restrict__ sumsq,
                   __bf16* __restrict__ R16) {
  extern __shared__ float aug[];       // [160][321] = 205,440 B (CDNA5 320KB WGP LDS)
  const int LD = 321;
  const int r = blockIdx.x;
  const int tid = threadIdx.x;
  const float* Qr = Q + (long)r * BSZ * BSZ;
  const float n = sqrtf(*sumsq);
  const float s = (fminf(n, CONSTRAINT_V) + EPS_N) / (n + EPS_N);

  for (int idx = tid; idx < BSZ * 2 * BSZ; idx += 256) {
    int i = idx / (2 * BSZ), j = idx % (2 * BSZ);
    float q = Qr[i * BSZ + (j % BSZ)] * s;
    float val;
    if (j < BSZ) val = ((i == j) ? 1.0f : 0.0f) - q;       // A = I - Q
    else         val = ((i == j - BSZ) ? 1.0f : 0.0f) + q; // B = I + Q
    aug[i * LD + j] = val;
  }
  __syncthreads();

  for (int kp = 0; kp < BSZ; ++kp) {
    float p = aug[kp * LD + kp];
    __syncthreads();
    for (int j = kp + tid; j < 2 * BSZ; j += 256)
      aug[kp * LD + j] = aug[kp * LD + j] / p;
    __syncthreads();
    if (tid < BSZ && tid != kp) {
      float f = aug[tid * LD + kp];
      for (int j = kp; j < 2 * BSZ; ++j)
        aug[tid * LD + j] -= f * aug[kp * LD + j];
    }
    __syncthreads();
  }
  // R = (I-Q)^-1 (I+Q) (commutes); padded row pitch BPAD, zeros beyond col 160
  for (int idx = tid; idx < BSZ * BPAD; idx += 256) {
    int c = idx / BPAD, d = idx % BPAD;
    R16[(long)r * BSZ * BPAD + idx] =
        (d < BSZ) ? (__bf16)aug[c * LD + BSZ + d] : (__bf16)0.0f;
  }
}

// ------------------------------------------------------- generic WMMA GEMM
// C[M,N] = A[M,K]*B[K,N]; bf16 in / f32 acc; 128x128x32 tiles; 8 waves
// (4 row x 2 col), each wave 32x64 = 2x4 WMMA tiles. Double-buffered LDS with
// register prefetch; B staged TRANSPOSED [n][k] so B-frags are K-contiguous.
// REQUIREMENT: B rows must be readable (and zero where unused) for 8-wide
// aligned vector loads across [tileN, tileN+BN); K % 32 == 0.
#define BM 128
#define BN 128
#define BK 32
#define LDT (BK + 8)   // 40 elems = 80B rows, 16B-aligned chunks

__global__ __launch_bounds__(256)
void gemm_bf16_kernel(const __bf16* __restrict__ A, int lda,
                      const __bf16* __restrict__ Bm, int ldb,
                      void* __restrict__ Cout, int ldc,
                      const float* __restrict__ bias,
                      int M, int N, int K,
                      int a_blk_stride, int b_blk_stride, int c_blk_stride,
                      int out_f32) {
  __shared__ alignas(16) __bf16 As[2][BM][LDT];
  __shared__ alignas(16) __bf16 Bs[2][BN][LDT];   // transposed: [n][k]

  const int tid = threadIdx.x;
  const int lane = tid & 31, wv = tid >> 5;
  const int wr = wv & 3, wc = wv >> 2;
  const int tileM = blockIdx.y * BM;
  const int tileN = blockIdx.x * BN;
  A  += (long)blockIdx.z * a_blk_stride;
  Bm += (long)blockIdx.z * b_blk_stride;
  const int m = lane & 15, khalf = lane >> 4;

  // staging: 2x A-chunks (8 bf16) + 2x B-chunks per thread
  const int a_r = tid >> 2, a_c = (tid & 3) * 8;    // rows a_r, a_r+64
  const int b_k = tid >> 4, b_n = (tid & 15) * 8;   // k-rows b_k, b_k+16

  union { U4 v; __bf16 h[8]; } ar[2], br[2];

  auto fetch = [&](int k0) {
#pragma unroll
    for (int i = 0; i < 2; ++i) {
      int gm = tileM + a_r + 64 * i;
      ar[i].v = zero4();
      if (gm < M) ar[i].v = *(const U4*)&A[(long)gm * lda + k0 + a_c];
    }
#pragma unroll
    for (int i = 0; i < 2; ++i) {
      int gk = k0 + b_k + 16 * i;
      br[i].v = *(const U4*)&Bm[(long)gk * ldb + tileN + b_n];
    }
  };
  auto stage = [&](int p) {
#pragma unroll
    for (int i = 0; i < 2; ++i)
      *(U4*)&As[p][a_r + 64 * i][a_c] = ar[i].v;
#pragma unroll
    for (int i = 0; i < 2; ++i)
#pragma unroll
      for (int e = 0; e < 8; ++e)
        Bs[p][b_n + e][b_k + 16 * i] = br[i].h[e];   // transpose into [n][k]
  };

  v8f acc[2][4];
#pragma unroll
  for (int ri = 0; ri < 2; ++ri)
#pragma unroll
    for (int ci = 0; ci < 4; ++ci) acc[ri][ci] = {};

  fetch(0);
  stage(0);
  __syncthreads();
  int p = 0;
  for (int k0 = 0; k0 < K; k0 += BK) {
    const bool more = (k0 + BK) < K;
    if (more) fetch(k0 + BK);          // global loads in flight during WMMAs

    v16bf af[2], bfg[4];
#pragma unroll
    for (int ri = 0; ri < 2; ++ri) {
      const __bf16* pa = &As[p][wr * 32 + ri * 16 + m][khalf * 16];
#pragma unroll
      for (int j = 0; j < 8; ++j) { af[ri][j] = pa[j]; af[ri][8 + j] = pa[8 + j]; }
    }
#pragma unroll
    for (int ci = 0; ci < 4; ++ci) {
      const __bf16* pb = &Bs[p][wc * 64 + ci * 16 + m][khalf * 16];
#pragma unroll
      for (int j = 0; j < 16; ++j) bfg[ci][j] = pb[j];
    }
#pragma unroll
    for (int ri = 0; ri < 2; ++ri)
#pragma unroll
      for (int ci = 0; ci < 4; ++ci)
        acc[ri][ci] = wmma_bf16(af[ri], bfg[ci], acc[ri][ci]);

    if (more) {
      stage(p ^ 1);
      __syncthreads();
      p ^= 1;
    }
  }

#pragma unroll
  for (int ri = 0; ri < 2; ++ri)
#pragma unroll
    for (int ci = 0; ci < 4; ++ci)
#pragma unroll
      for (int v = 0; v < 8; ++v) {
        int gm = tileM + wr * 32 + ri * 16 + v + khalf * 8;   // ISA C/D layout
        int gn = tileN + wc * 64 + ci * 16 + m;
        if (gm < M && gn < N) {
          long co = (long)blockIdx.z * c_blk_stride + (long)gm * ldc + gn;
          float val = acc[ri][ci][v];
          if (out_f32) ((float*)Cout)[co] = val + (bias ? bias[gn] : 0.0f);
          else         ((__bf16*)Cout)[co] = (__bf16)val;
        }
      }
}

// ------------------------------------------------------------ fused attention
// grid (S/64, heads, B), 128 thr (4 waves); each wave owns 16 q-rows.
// Klds natural [key][d]; Vlds transposed [d][key] => both B-frags K-contiguous.
__global__ __launch_bounds__(128)
void attention_kernel(const __bf16* __restrict__ qr, const __bf16* __restrict__ kr,
                      const __bf16* __restrict__ vr, __bf16* __restrict__ attn) {
  const int b = blockIdx.z, h = blockIdx.y, qt = blockIdx.x;
  const int tid = threadIdx.x, w = tid >> 5, lane = tid & 31;
  const int m = lane & 15, khalf = lane >> 4;
  const float scale = 0.125f; // 1/sqrt(64)

  __shared__ alignas(16) __bf16 Klds[80][72];        // [key][d], pad 77->80
  __shared__ alignas(16) __bf16 Vlds[64][104];       // [d][key], pad 77->96
  __shared__ float  Slds[4][16][82];
  __shared__ alignas(16) __bf16 Plds[4][16][104];

  for (int idx = tid; idx < 80 * 8; idx += 128) {    // K: 80 rows x 8 chunks
    int key = idx >> 3, cc = (idx & 7) * 8;
    U4 val = zero4();
    if (key < SEQE) val = *(const U4*)&kr[((long)b * SEQE + key) * H_DIM + h * HD + cc];
    *(U4*)&Klds[key][cc] = val;
  }
  for (int idx = tid; idx < 96 * 8; idx += 128) {    // V: transpose into [d][key]
    int key = idx >> 3, cc = (idx & 7) * 8;
    union { U4 v; __bf16 h[8]; } t;
    t.v = zero4();
    if (key < SEQE) t.v = *(const U4*)&vr[((long)b * SEQE + key) * H_DIM + h * HD + cc];
#pragma unroll
    for (int e = 0; e < 8; ++e) Vlds[cc + e][key] = t.h[e];
  }
  __syncthreads();

  const long qrow0 = (long)b * SEQ + qt * 64 + w * 16;
  const __bf16* pq = &qr[(qrow0 + m) * H_DIM + h * HD];
  v16bf aq[2];
#pragma unroll
  for (int ks = 0; ks < 2; ++ks)
#pragma unroll
    for (int j = 0; j < 8; ++j) {
      aq[ks][j]     = pq[ks * 32 + khalf * 16 + j];
      aq[ks][8 + j] = pq[ks * 32 + khalf * 16 + 8 + j];
    }

  // scores = Q @ K^T (5 key tiles x 2 k-steps)
#pragma unroll
  for (int nt = 0; nt < 5; ++nt) {
    v8f acc = {};
#pragma unroll
    for (int ks = 0; ks < 2; ++ks) {
      v16bf bk;
      const __bf16* pk = &Klds[nt * 16 + m][ks * 32 + khalf * 16];
#pragma unroll
      for (int j = 0; j < 16; ++j) bk[j] = pk[j];
      acc = wmma_bf16(aq[ks], bk, acc);
    }
#pragma unroll
    for (int v = 0; v < 8; ++v)
      Slds[w][v + khalf * 8][nt * 16 + m] = acc[v] * scale;
  }
  __syncthreads();

  // softmax over 77 valid keys
  if (lane < 16) {
    float mx = -1e30f;
    for (int j = 0; j < SEQE; ++j) mx = fmaxf(mx, Slds[w][lane][j]);
    float sum = 0.0f;
    for (int j = 0; j < SEQE; ++j) {
      float e = __expf(Slds[w][lane][j] - mx);
      Slds[w][lane][j] = e;
      sum += e;
    }
    float inv = 1.0f / sum;
    for (int j = 0; j < 96; ++j)
      Plds[w][lane][j] = (j < SEQE) ? (__bf16)(Slds[w][lane][j] * inv)
                                    : (__bf16)0.0f;
  }
  __syncthreads();

  // out = P @ V (4 d-tiles x 3 k-steps over padded 96 keys)
#pragma unroll
  for (int ct = 0; ct < 4; ++ct) {
    v8f acc = {};
#pragma unroll
    for (int ks = 0; ks < 3; ++ks) {
      v16bf ap, bv;
      const __bf16* pp = &Plds[w][m][ks * 32 + khalf * 16];
      const __bf16* pv = &Vlds[ct * 16 + m][ks * 32 + khalf * 16];
#pragma unroll
      for (int j = 0; j < 8; ++j) { ap[j] = pp[j]; ap[8 + j] = pp[8 + j]; }
#pragma unroll
      for (int j = 0; j < 16; ++j) bv[j] = pv[j];
      acc = wmma_bf16(ap, bv, acc);
    }
#pragma unroll
    for (int v = 0; v < 8; ++v)
      attn[(qrow0 + v + khalf * 8) * H_DIM + h * HD + ct * 16 + m] = (__bf16)acc[v];
  }
}

// ------------------------------------------------------------------- driver
static inline int ceil_div(int a, int b) { return (a + b - 1) / b; }

extern "C" void kernel_launch(void* const* d_in, const int* in_sizes, int n_in,
                              void* d_out, int out_size, void* d_ws, size_t ws_size,
                              hipStream_t stream) {
  (void)in_sizes; (void)n_in; (void)out_size; (void)ws_size;
  const float* hs  = (const float*)d_in[0];
  const float* ehs = (const float*)d_in[1];
  const float* WQb = (const float*)d_in[2];
  const float* WKb = (const float*)d_in[3];
  const float* WVb = (const float*)d_in[4];
  const float* Wq  = (const float*)d_in[5];
  const float* Wk  = (const float*)d_in[6];
  const float* Wv  = (const float*)d_in[7];
  const float* Wo  = (const float*)d_in[8];
  const float* bo  = (const float*)d_in[9];
  float* out = (float*)d_out;

  const int MQ = BATCH * SEQ;                 // 16384
  const int MK = BATCH * SEQE;                // 1232
  const int BLK_ELEMS  = NBLK * BSZ * BSZ;    // 204800 (fp32 skew blocks)
  const int BLKP_ELEMS = NBLK * BSZ * BPAD;   // 327680 (padded bf16 R blocks)

  char* base = (char*)d_ws;
  size_t off = 0;
  auto alloc = [&](size_t bytes) -> void* {
    off = (off + 255) & ~(size_t)255;
    void* p = base + off;
    off += bytes;
    return p;
  };
  float*  norms  = (float*)alloc(3 * sizeof(float));
  float*  Qsk    = (float*)alloc((size_t)3 * BLK_ELEMS * 4);
  __bf16* Rb16   = (__bf16*)alloc((size_t)3 * BLKP_ELEMS * 2);
  __bf16* hs16   = (__bf16*)alloc((size_t)MQ * H_DIM * 2);
  __bf16* ehs16  = (__bf16*)alloc((size_t)MK * CENC * 2);
  __bf16* wq16   = (__bf16*)alloc((size_t)H_DIM * H_DIM * 2);
  __bf16* wk16   = (__bf16*)alloc((size_t)CENC * H_DIM * 2);
  __bf16* wv16   = (__bf16*)alloc((size_t)CENC * H_DIM * 2);
  __bf16* wo16   = (__bf16*)alloc((size_t)H_DIM * H_DIM * 2);
  __bf16* wqrot  = (__bf16*)alloc((size_t)H_DIM * H_DIM * 2);   // Wq @ blockdiag(R_Q)
  __bf16* wkrot  = (__bf16*)alloc((size_t)CENC * H_DIM * 2);
  __bf16* wvrot  = (__bf16*)alloc((size_t)CENC * H_DIM * 2);
  __bf16* qr16   = (__bf16*)alloc((size_t)MQ * H_DIM * 2);
  __bf16* kr16   = (__bf16*)alloc((size_t)MK * H_DIM * 2);
  __bf16* vr16   = (__bf16*)alloc((size_t)MK * H_DIM * 2);
  __bf16* attn16 = hs16;   // hs16 dead after q projection; reuse

  // 1) Cayley prep
  zero_f32_kernel<<<1, 32, 0, stream>>>(norms, 3);
  skew_norm_kernel<<<NBLK, 256, 0, stream>>>(WQb, Qsk + 0 * BLK_ELEMS, norms + 0);
  skew_norm_kernel<<<NBLK, 256, 0, stream>>>(WKb, Qsk + 1 * BLK_ELEMS, norms + 1);
  skew_norm_kernel<<<NBLK, 256, 0, stream>>>(WVb, Qsk + 2 * BLK_ELEMS, norms + 2);

  // 2) bf16 conversions (vectorized x4)
  cvt_f32_bf16_kernel<<<2048, 256, 0, stream>>>(hs,  hs16,  MQ * H_DIM / 4);
  cvt_f32_bf16_kernel<<<512, 256, 0, stream>>>(ehs, ehs16, MK * CENC / 4);
  cvt_f32_bf16_kernel<<<512, 256, 0, stream>>>(Wq,  wq16,  H_DIM * H_DIM / 4);
  cvt_f32_bf16_kernel<<<512, 256, 0, stream>>>(Wk,  wk16,  CENC * H_DIM / 4);
  cvt_f32_bf16_kernel<<<512, 256, 0, stream>>>(Wv,  wv16,  CENC * H_DIM / 4);
  cvt_f32_bf16_kernel<<<512, 256, 0, stream>>>(Wo,  wo16,  H_DIM * H_DIM / 4);

  // 3) Cayley Gauss-Jordan (205KB dynamic LDS per workgroup), padded R blocks
  const size_t gj_lds = (size_t)BSZ * 321 * sizeof(float);
  cayley_kernel<<<NBLK, 256, gj_lds, stream>>>(Qsk + 0 * BLK_ELEMS, norms + 0, Rb16 + 0 * BLKP_ELEMS);
  cayley_kernel<<<NBLK, 256, gj_lds, stream>>>(Qsk + 1 * BLK_ELEMS, norms + 1, Rb16 + 1 * BLKP_ELEMS);
  cayley_kernel<<<NBLK, 256, gj_lds, stream>>>(Qsk + 2 * BLK_ELEMS, norms + 2, Rb16 + 2 * BLKP_ELEMS);

  // 4) fold OFT rotations into the projection weights:
  //    (x @ W) @ blockdiag(R) == x @ (W @ blockdiag(R))   [~0.5 GFLOP each,
  //    saves the 170MB per-token rotate passes entirely]
  gemm_bf16_kernel<<<dim3(ceil_div(BSZ, BN), ceil_div(H_DIM, BM), NBLK), 256, 0, stream>>>(
      wq16, H_DIM, Rb16 + 0 * BLKP_ELEMS, BPAD, wqrot, H_DIM, nullptr,
      H_DIM, BSZ, BSZ, BSZ, BSZ * BPAD, BSZ, 0);
  gemm_bf16_kernel<<<dim3(ceil_div(BSZ, BN), ceil_div(CENC, BM), NBLK), 256, 0, stream>>>(
      wk16, H_DIM, Rb16 + 1 * BLKP_ELEMS, BPAD, wkrot, H_DIM, nullptr,
      CENC, BSZ, BSZ, BSZ, BSZ * BPAD, BSZ, 0);
  gemm_bf16_kernel<<<dim3(ceil_div(BSZ, BN), ceil_div(CENC, BM), NBLK), 256, 0, stream>>>(
      wv16, H_DIM, Rb16 + 2 * BLKP_ELEMS, BPAD, wvrot, H_DIM, nullptr,
      CENC, BSZ, BSZ, BSZ, BSZ * BPAD, BSZ, 0);

  // 5) projections produce rotated q/k/v directly
  gemm_bf16_kernel<<<dim3(ceil_div(H_DIM, BN), ceil_div(MQ, BM), 1), 256, 0, stream>>>(
      hs16, H_DIM, wqrot, H_DIM, qr16, H_DIM, nullptr, MQ, H_DIM, H_DIM, 0, 0, 0, 0);
  gemm_bf16_kernel<<<dim3(ceil_div(H_DIM, BN), ceil_div(MK, BM), 1), 256, 0, stream>>>(
      ehs16, CENC, wkrot, H_DIM, kr16, H_DIM, nullptr, MK, H_DIM, CENC, 0, 0, 0, 0);
  gemm_bf16_kernel<<<dim3(ceil_div(H_DIM, BN), ceil_div(MK, BM), 1), 256, 0, stream>>>(
      ehs16, CENC, wvrot, H_DIM, vr16, H_DIM, nullptr, MK, H_DIM, CENC, 0, 0, 0, 0);

  // 6) fused attention
  attention_kernel<<<dim3(SEQ / 64, NHEADS, BATCH), 128, 0, stream>>>(
      qr16, kr16, vr16, attn16);

  // 7) output projection + bias (fp32 out)
  gemm_bf16_kernel<<<dim3(ceil_div(H_DIM, BN), ceil_div(MQ, BM), 1), 256, 0, stream>>>(
      attn16, H_DIM, wo16, H_DIM, out, H_DIM, bo, MQ, H_DIM, H_DIM, 0, 0, 0, 1);
}
